// NodeRepresentation_77996606095380
// MI455X (gfx1250) — matrix operand order repaired and checked by reading, not compiled, not run on hardware
//
#include <hip/hip_runtime.h>
#include <hip/hip_bf16.h>

// ---------------------------------------------------------------------------
// Model dims
#define N_NODES  10000
#define N_EDGES  80000
#define E_TOT    90000          // + self loops
#define NG       512
#define BSZ      512
#define F_IN     75
#define UDIM     256
#define OUTC     128
#define DIM_GEXP 697
#define DIM_METH 808
#define MUT_W    3489
#define SLOPE    0.2f
#define BN_EPS   1e-5f

typedef __attribute__((ext_vector_type(16))) __bf16 bf16x16;
typedef __attribute__((ext_vector_type(8)))  float  f32x8;
typedef __attribute__((ext_vector_type(4)))  unsigned int u32x4;
typedef __attribute__((ext_vector_type(4)))  int    i32x4;
typedef __attribute__((ext_vector_type(8)))  int    i32x8;

__device__ __forceinline__ float lrelu(float v) { return v > 0.f ? v : SLOPE * v; }

__device__ float atomicMaxF(float* addr, float val) {
    int* ia = (int*)addr;
    int old = __float_as_int(*addr);
    while (__int_as_float(old) < val) {
        int assumed = old;
        old = atomicCAS(ia, assumed, __float_as_int(val));
        if (old == assumed) break;
    }
    return __int_as_float(old);
}

// ---------------------------------------------------------------------------
__global__ void fill_f32(float* p, float v, long n) {
    long i = blockIdx.x * (long)blockDim.x + threadIdx.x;
    long s = (long)gridDim.x * blockDim.x;
    for (; i < n; i += s) p[i] = v;
}

// ---------------------------------------------------------------------------
// Shared GEMM geometry: block = 256 threads = 8 waves (wave32),
// block tile 128x64, each wave computes a 32x32 region (2x2 WMMA tiles).
#define BM 128
#define BN 64
#define BK 32
#define LDS_STRIDE 40   // elements per LDS tile row (pad kills bank conflicts)

// GEMM: Y[M,N] = act( A[M,K] @ W[N,K]^T + bias[N] )
// bf16 tiles staged in LDS with vectorized fast path, v_wmma_f32_16x16x32_bf16.
__global__ __launch_bounds__(256) void gemm_bias_act(
    const float* __restrict__ A, const float* __restrict__ W,
    const float* __restrict__ bias, float* __restrict__ Y,
    int M, int N, int K, int act)
{
    __shared__ __bf16 sA[BM * LDS_STRIDE];
    __shared__ __bf16 sB[BN * LDS_STRIDE];

    const int tid  = threadIdx.x;
    const int lane = tid & 31;
    const int w    = tid >> 5;
    const int wm   = w >> 1;          // 0..3
    const int wn   = w & 1;           // 0..1
    const int m16  = lane & 15;
    const int kh   = lane >> 4;       // 0/1 half of the wave
    const int row0 = blockIdx.x * BM;
    const int col0 = blockIdx.y * BN;
    const bool k4  = (K & 3) == 0;

    f32x8 acc[2][2] = {};

    for (int k0 = 0; k0 < K; k0 += BK) {
        const bool fullK = (k0 + BK) <= K;
        if (fullK && (k0 + BK) < K) {
            __builtin_prefetch(A + (long)(row0 + (tid >> 1)) * K + k0 + BK, 0, 1);
            if (tid < BN) __builtin_prefetch(W + (long)(col0 + tid) * K + k0 + BK, 0, 1);
        }
        // ---- stage A ----
        if (k4 && fullK && (row0 + BM) <= M) {
            for (int i4 = tid; i4 < BM * BK / 4; i4 += 256) {
                int r = i4 >> 3, kg = (i4 & 7) * 4;
                const float4 f = *(const float4*)(A + (long)(row0 + r) * K + k0 + kg);
                union { __bf16 h[4]; unsigned long long u; } pk;
                pk.h[0] = (__bf16)f.x; pk.h[1] = (__bf16)f.y;
                pk.h[2] = (__bf16)f.z; pk.h[3] = (__bf16)f.w;
                *(unsigned long long*)&sA[r * LDS_STRIDE + kg] = pk.u;
            }
        } else {
            float vals[16];
#pragma unroll
            for (int j = 0; j < 16; ++j) {
                int i = tid + j * 256;
                int r = i >> 5, kk = i & 31;
                int gr = row0 + r, gk = k0 + kk;
                vals[j] = (gr < M && gk < K) ? A[(long)gr * K + gk] : 0.f;
            }
#pragma unroll
            for (int j = 0; j < 16; ++j) {
                int i = tid + j * 256;
                sA[(i >> 5) * LDS_STRIDE + (i & 31)] = (__bf16)vals[j];
            }
        }
        // ---- stage W ----
        if (k4 && fullK && (col0 + BN) <= N) {
            for (int i4 = tid; i4 < BN * BK / 4; i4 += 256) {
                int r = i4 >> 3, kg = (i4 & 7) * 4;
                const float4 f = *(const float4*)(W + (long)(col0 + r) * K + k0 + kg);
                union { __bf16 h[4]; unsigned long long u; } pk;
                pk.h[0] = (__bf16)f.x; pk.h[1] = (__bf16)f.y;
                pk.h[2] = (__bf16)f.z; pk.h[3] = (__bf16)f.w;
                *(unsigned long long*)&sB[r * LDS_STRIDE + kg] = pk.u;
            }
        } else {
            float vals[8];
#pragma unroll
            for (int j = 0; j < 8; ++j) {
                int i = tid + j * 256;
                int r = i >> 5, kk = i & 31;
                int gn = col0 + r, gk = k0 + kk;
                vals[j] = (gn < N && gk < K) ? W[(long)gn * K + gk] : 0.f;
            }
#pragma unroll
            for (int j = 0; j < 8; ++j) {
                int i = tid + j * 256;
                sB[(i >> 5) * LDS_STRIDE + (i & 31)] = (__bf16)vals[j];
            }
        }
        __syncthreads();

        bf16x16 afr[2], bfr[2];
#pragma unroll
        for (int tm = 0; tm < 2; ++tm) {
            const __bf16* base = &sA[(wm * 32 + tm * 16 + m16) * LDS_STRIDE];
#pragma unroll
            for (int j = 0; j < 8; ++j) afr[tm][j]     = base[kh * 8 + j];
#pragma unroll
            for (int j = 0; j < 8; ++j) afr[tm][8 + j] = base[16 + kh * 8 + j];
        }
#pragma unroll
        for (int tn = 0; tn < 2; ++tn) {
            const __bf16* base = &sB[(wn * 32 + tn * 16 + m16) * LDS_STRIDE];
#pragma unroll
            for (int j = 0; j < 16; ++j) bfr[tn][j] = base[(j >> 3) * 16 + kh * 8 + (j & 7)];
        }
#pragma unroll
        for (int tm = 0; tm < 2; ++tm)
#pragma unroll
            for (int tn = 0; tn < 2; ++tn)
                acc[tm][tn] = __builtin_amdgcn_wmma_f32_16x16x32_bf16(
                    false, afr[tm], false, bfr[tn], (short)0, acc[tm][tn],
                    false, false);
        __syncthreads();
    }

#pragma unroll
    for (int tm = 0; tm < 2; ++tm) {
#pragma unroll
        for (int tn = 0; tn < 2; ++tn) {
            int col = col0 + wn * 32 + tn * 16 + m16;
#pragma unroll
            for (int v = 0; v < 8; ++v) {
                int row = row0 + wm * 32 + tm * 16 + v + 8 * kh;
                if (row < M && col < N) {
                    float o = acc[tm][tn][v] + (bias ? bias[col] : 0.f);
                    if (act) o = lrelu(o);
                    Y[(long)row * N + col] = o;
                }
            }
        }
    }
}

// ---------------------------------------------------------------------------
// TDM-staged GEMM, double-buffered: the Tensor Data Mover pulls tile k+1 into
// the alternate LDS buffer while all 8 waves run WMMA on tile k.
// TDM ops from one wave complete in order, so s_wait_tensorcnt 2 guarantees
// the current tile's two loads are done while the next tile's are in flight.
// D# per ISA 8.3/8.4: count=1, type=2, data_size=4B, LDS pad 8dw per 32dw
// (pad_interval=4, pad_amount=7) -> 40-dword row stride matching fragments.
__device__ __forceinline__ int rfl(unsigned v) {
    return __builtin_amdgcn_readfirstlane((int)v);
}

__device__ __forceinline__ void tdm_load_2d(
    const float* gbase, unsigned ldsOff, int td0, int td1,
    int tile0, int tile1, int strideElems)
{
    unsigned long long ga = (unsigned long long)(const void*)gbase;
    u32x4 g0;
    g0[0] = 1u;                                    // count=1, user mode
    g0[1] = (unsigned)rfl(ldsOff);                 // lds_addr (bytes)
    g0[2] = (unsigned)rfl((unsigned)ga);           // global_addr[31:0]
    g0[3] = (unsigned)rfl(((unsigned)(ga >> 32) & 0x01FFFFFFu) | 0x80000000u); // [56:32] | type=2
    i32x8 g1;
    g1[0] = 0x0F120000;                            // data_size=4B, pad_en, int=32dw, amt=8dw
    g1[1] = rfl((unsigned)(td0 & 0xFFFF) << 16);
    g1[2] = rfl(((unsigned)td0 >> 16) | ((unsigned)(td1 & 0xFFFF) << 16));
    g1[3] = rfl(((unsigned)td1 >> 16) | ((unsigned)(tile0 & 0xFFFF) << 16));
    g1[4] = rfl((unsigned)(tile1 & 0xFFFF));
    g1[5] = rfl((unsigned)strideElems);
    g1[6] = 0;
    g1[7] = 0;
    i32x4 gz = {};
#if __clang_major__ >= 23
    i32x8 gz8 = {};
    __builtin_amdgcn_tensor_load_to_lds(g0, g1, gz, gz, gz8, 0);
#else
    __builtin_amdgcn_tensor_load_to_lds(g0, g1, gz, gz, 0);
#endif
}

__global__ __launch_bounds__(256) void gemm_tdm_bias_act(
    const float* __restrict__ A, const float* __restrict__ W,
    const float* __restrict__ bias, float* __restrict__ Y,
    int M, int N, int K, int act)
{
    __shared__ float sAf[2][BM * LDS_STRIDE];   // 2 x 20 KB
    __shared__ float sBf[2][BN * LDS_STRIDE];   // 2 x 10 KB

    const int tid  = threadIdx.x;
    const int lane = tid & 31;
    const int w    = tid >> 5;
    const int wm   = w >> 1;
    const int wn   = w & 1;
    const int m16  = lane & 15;
    const int kh   = lane >> 4;
    const int row0 = blockIdx.x * BM;
    const int col0 = blockIdx.y * BN;

    const unsigned ldsA0 = (unsigned)(unsigned long long)(void*)&sAf[0][0];
    const unsigned ldsA1 = (unsigned)(unsigned long long)(void*)&sAf[1][0];
    const unsigned ldsB0 = (unsigned)(unsigned long long)(void*)&sBf[0][0];
    const unsigned ldsB1 = (unsigned)(unsigned long long)(void*)&sBf[1][0];

    f32x8 acc[2][2] = {};
    const int nk = (K + BK - 1) / BK;

    if (tid < 32) {   // wave 0 drives the DMA
        tdm_load_2d(A + (long)row0 * K, ldsA0, K, M - row0, BK, BM, K);
        tdm_load_2d(W + (long)col0 * K, ldsB0, K, N - col0, BK, BN, K);
    }

    for (int ki = 0; ki < nk; ++ki) {
        const int cur = ki & 1;
        if (tid < 32) {
            if (ki + 1 < nk) {
                const int k1 = (ki + 1) * BK;
                tdm_load_2d(A + (long)row0 * K + k1, cur ? ldsA0 : ldsA1,
                            K - k1, M - row0, BK, BM, K);
                tdm_load_2d(W + (long)col0 * K + k1, cur ? ldsB0 : ldsB1,
                            K - k1, N - col0, BK, BN, K);
                __builtin_amdgcn_s_wait_tensorcnt(2);  // tile ki done, ki+1 in flight
            } else {
                __builtin_amdgcn_s_wait_tensorcnt(0);
            }
        }
        __syncthreads();

        const float* cA = &sAf[cur][0];
        const float* cB = &sBf[cur][0];
        bf16x16 afr[2], bfr[2];
#pragma unroll
        for (int tm = 0; tm < 2; ++tm) {
            const float* base = cA + (wm * 32 + tm * 16 + m16) * LDS_STRIDE;
#pragma unroll
            for (int j = 0; j < 8; ++j) afr[tm][j]     = (__bf16)base[kh * 8 + j];
#pragma unroll
            for (int j = 0; j < 8; ++j) afr[tm][8 + j] = (__bf16)base[16 + kh * 8 + j];
        }
#pragma unroll
        for (int tn = 0; tn < 2; ++tn) {
            const float* base = cB + (wn * 32 + tn * 16 + m16) * LDS_STRIDE;
#pragma unroll
            for (int j = 0; j < 16; ++j) bfr[tn][j] = (__bf16)base[(j >> 3) * 16 + kh * 8 + (j & 7)];
        }
#pragma unroll
        for (int tm = 0; tm < 2; ++tm)
#pragma unroll
            for (int tn = 0; tn < 2; ++tn)
                acc[tm][tn] = __builtin_amdgcn_wmma_f32_16x16x32_bf16(
                    false, afr[tm], false, bfr[tn], (short)0, acc[tm][tn],
                    false, false);
        __syncthreads();  // protect buffer `cur` before next iteration's DMA
    }

#pragma unroll
    for (int tm = 0; tm < 2; ++tm) {
#pragma unroll
        for (int tn = 0; tn < 2; ++tn) {
            int col = col0 + wn * 32 + tn * 16 + m16;
#pragma unroll
            for (int v = 0; v < 8; ++v) {
                int row = row0 + wm * 32 + tm * 16 + v + 8 * kh;
                if (row < M && col < N) {
                    float o = acc[tm][tn][v] + (bias ? bias[col] : 0.f);
                    if (act) o = lrelu(o);
                    Y[(long)row * N + col] = o;
                }
            }
        }
    }
}

// ---------------------------------------------------------------------------
// GATv2 edge kernels.  Edge e<N_EDGES uses edge_index, else self loop e-N_EDGES.
__global__ void edge_logits_max(
    const float* __restrict__ xl, const float* __restrict__ xr,
    const float* __restrict__ att, const int* __restrict__ esrc,
    const int* __restrict__ edst, int Emain, int Etot, int H, int C,
    float* __restrict__ logit, float* __restrict__ segmax)
{
    long gw = (blockIdx.x * (long)blockDim.x + threadIdx.x) >> 5;
    int lane = threadIdx.x & 31;
    long e = gw / H;
    int  h = (int)(gw % H);
    if (e >= Etot) return;
    int s, d;
    if (e < Emain) { s = esrc[e]; d = edst[e]; } else { s = d = (int)(e - Emain); }
    const float* pl = xl + (long)s * H * C + (long)h * C;
    const float* pr = xr + (long)d * H * C + (long)h * C;
    const float* pa = att + (long)h * C;
    float sum = 0.f;
    for (int c = lane; c < C; c += 32) sum += lrelu(pl[c] + pr[c]) * pa[c];
#pragma unroll
    for (int off = 16; off; off >>= 1) sum += __shfl_xor(sum, off, 32);
    if (lane == 0) {
        logit[e * H + h] = sum;
        atomicMaxF(&segmax[d * H + h], sum);
    }
}

__global__ void edge_exp_den(
    const float* __restrict__ logit, const float* __restrict__ segmax,
    const int* __restrict__ edst, int Emain, int Etot, int H,
    float* __restrict__ ex, float* __restrict__ den)
{
    long i = blockIdx.x * (long)blockDim.x + threadIdx.x;
    if (i >= (long)Etot * H) return;
    long e = i / H;
    int  h = (int)(i % H);
    int  d = (e < Emain) ? edst[e] : (int)(e - Emain);
    float v = __expf(logit[i] - segmax[d * H + h]);
    ex[i] = v;
    atomicAdd(&den[d * H + h], v);
}

// wave per (edge, head): alpha computed once, lanes stream C channels
__global__ void edge_scatter(
    const float* __restrict__ xl, const float* __restrict__ ex,
    const float* __restrict__ den, const int* __restrict__ esrc,
    const int* __restrict__ edst, int Emain, int Etot, int H, int C,
    float* __restrict__ headout)
{
    long gw = (blockIdx.x * (long)blockDim.x + threadIdx.x) >> 5;
    int lane = threadIdx.x & 31;
    long e = gw / H;
    int  h = (int)(gw % H);
    if (e >= Etot) return;
    int s, d;
    if (e < Emain) { s = esrc[e]; d = edst[e]; } else { s = d = (int)(e - Emain); }
    float alpha = ex[e * H + h] / den[d * H + h];
    const float* px = xl + (long)s * H * C + (long)h * C;
    float* po = headout + (long)d * H * C + (long)h * C;
    for (int c = lane; c < C; c += 32)
        atomicAdd(&po[c], px[c] * alpha);
}

__global__ void head_mean_bias(
    const float* __restrict__ headout, const float* __restrict__ bias,
    float* __restrict__ out, int Nn, int H, int C)
{
    long total = (long)Nn * C;
    long stride = (long)gridDim.x * blockDim.x;
    for (long i = blockIdx.x * (long)blockDim.x + threadIdx.x; i < total; i += stride) {
        long n = i / C;
        int  c = (int)(i % C);
        float s = 0.f;
        for (int h = 0; h < H; ++h) s += headout[n * H * C + (long)h * C + c];
        out[i] = s / H + bias[c];
    }
}

// ---------------------------------------------------------------------------
// BatchNorm1d: per-column batch stats (biased var).  One block per column.
__global__ __launch_bounds__(256) void col_stats(
    const float* __restrict__ X, int M, int N, int lrelu_in,
    float* __restrict__ mean, float* __restrict__ var)
{
    int j = blockIdx.x;
    __shared__ float ss[256], sq[256];
    float s = 0.f, q = 0.f;
    for (int r = threadIdx.x; r < M; r += 256) {
        float v = X[(long)r * N + j];
        if (lrelu_in) v = lrelu(v);
        s += v; q += v * v;
    }
    ss[threadIdx.x] = s; sq[threadIdx.x] = q;
    __syncthreads();
    for (int o = 128; o; o >>= 1) {
        if (threadIdx.x < o) { ss[threadIdx.x] += ss[threadIdx.x + o];
                               sq[threadIdx.x] += sq[threadIdx.x + o]; }
        __syncthreads();
    }
    if (threadIdx.x == 0) {
        float m = ss[0] / M;
        mean[j] = m;
        var[j]  = sq[0] / M - m * m;
    }
}

__global__ void bn_apply(
    const float* __restrict__ X, const float* __restrict__ res,
    const float* __restrict__ mean, const float* __restrict__ var,
    const float* __restrict__ g, const float* __restrict__ b,
    float* __restrict__ Y, long M, int N, int lrelu_in, int lrelu_out)
{
    long total = M * N;
    long stride = (long)gridDim.x * blockDim.x;
    for (long i = blockIdx.x * (long)blockDim.x + threadIdx.x; i < total; i += stride) {
        int j = (int)(i % N);
        float v = X[i];
        if (lrelu_in) v = lrelu(v);
        v = (v - mean[j]) * rsqrtf(var[j] + BN_EPS) * g[j] + b[j];
        if (lrelu_out) v = lrelu(v);
        if (res) v += res[i];
        Y[i] = v;
    }
}

// ---------------------------------------------------------------------------
// conv1 (Ci=1) scalar path, BN2d stats/apply, maxpool3
__global__ void conv1d_k(
    const float* __restrict__ X, const float* __restrict__ Wt,
    const float* __restrict__ bias, float* __restrict__ Y,
    int B, int Ci, int Wi, int Co, int Kw, int stride, int Wo)
{
    long total = (long)B * Co * Wo;
    long gstride = (long)gridDim.x * blockDim.x;
    for (long i = blockIdx.x * (long)blockDim.x + threadIdx.x; i < total; i += gstride) {
        int  ox = (int)(i % Wo);
        long t  = i / Wo;
        int  co = (int)(t % Co);
        int  b  = (int)(t / Co);
        float s = bias[co];
        int x0 = ox * stride;
        for (int ci = 0; ci < Ci; ++ci) {
            const float* xp = X + ((long)b * Ci + ci) * Wi + x0;
            const float* wp = Wt + ((long)co * Ci + ci) * Kw;
            for (int k = 0; k < Kw; ++k) s += xp[k] * wp[k];
        }
        Y[i] = s;
    }
}

__global__ __launch_bounds__(256) void chan_stats(
    const float* __restrict__ X, int B, int Cc, int W,
    float* __restrict__ mean, float* __restrict__ var)
{
    int c = blockIdx.x;
    __shared__ float ss[256], sq[256];
    float s = 0.f, q = 0.f;
    long n = (long)B * W;
    for (long i = threadIdx.x; i < n; i += 256) {
        long b = i / W;
        int  w = (int)(i % W);
        float v = X[(b * Cc + c) * (long)W + w];
        s += v; q += v * v;
    }
    ss[threadIdx.x] = s; sq[threadIdx.x] = q;
    __syncthreads();
    for (int o = 128; o; o >>= 1) {
        if (threadIdx.x < o) { ss[threadIdx.x] += ss[threadIdx.x + o];
                               sq[threadIdx.x] += sq[threadIdx.x + o]; }
        __syncthreads();
    }
    if (threadIdx.x == 0) {
        float m = ss[0] / n;
        mean[c] = m;
        var[c]  = sq[0] / n - m * m;
    }
}

__global__ void bn2d_apply_lrelu(
    const float* __restrict__ X, const float* __restrict__ mean,
    const float* __restrict__ var, const float* __restrict__ g,
    const float* __restrict__ b, float* __restrict__ Y,
    int B, int Cc, int W)
{
    long total = (long)B * Cc * W;
    long stride = (long)gridDim.x * blockDim.x;
    for (long i = blockIdx.x * (long)blockDim.x + threadIdx.x; i < total; i += stride) {
        int c = (int)((i / W) % Cc);
        float v = (X[i] - mean[c]) * rsqrtf(var[c] + BN_EPS) * g[c] + b[c];
        Y[i] = lrelu(v);
    }
}

__global__ void maxpool3_k(
    const float* __restrict__ X, float* __restrict__ Y,
    int BC, int Wi, int Wo)
{
    long total = (long)BC * Wo;
    long stride = (long)gridDim.x * blockDim.x;
    for (long i = blockIdx.x * (long)blockDim.x + threadIdx.x; i < total; i += stride) {
        int  ox = (int)(i % Wo);
        long bc = i / Wo;
        const float* xp = X + bc * Wi + ox * 3;
        Y[i] = fmaxf(xp[0], fmaxf(xp[1], xp[2]));
    }
}

// ---------------------------------------------------------------------------
// conv2 as implicit GEMM: im2col [B*Wo, Ci*Kw], channel-last output [B*Wo, Co]
__global__ void im2col_c2(
    const float* __restrict__ X, float* __restrict__ IM,
    int B, int Ci, int Wi, int Kw, int stride, int Wo)
{
    int CK = Ci * Kw;
    long total = (long)B * Wo * CK;
    long gstride = (long)gridDim.x * blockDim.x;
    for (long i = blockIdx.x * (long)blockDim.x + threadIdx.x; i < total; i += gstride) {
        int  j  = (int)(i % CK);
        long r  = i / CK;
        int  ox = (int)(r % Wo);
        int  b  = (int)(r / Wo);
        int ci = j / Kw, k = j % Kw;
        IM[i] = X[((long)b * Ci + ci) * Wi + ox * stride + k];
    }
}

// pool over width from channel-last [B, Wi, C] into flatten layout [B, C*Wo]
__global__ void maxpool3_cl(
    const float* __restrict__ Ycl, float* __restrict__ P,
    int B, int Cc, int Wi, int Wo)
{
    long total = (long)B * Cc * Wo;
    long stride = (long)gridDim.x * blockDim.x;
    for (long i = blockIdx.x * (long)blockDim.x + threadIdx.x; i < total; i += stride) {
        int  ox = (int)(i % Wo);
        long t  = i / Wo;
        int  c  = (int)(t % Cc);
        int  b  = (int)(t / Cc);
        const float* yp = Ycl + ((long)b * Wi + ox * 3) * Cc + c;
        float m = fmaxf(yp[0], fmaxf(yp[Cc], yp[2 * Cc]));
        P[(long)b * Cc * Wo + (long)c * Wo + ox] = m;
    }
}

// ---------------------------------------------------------------------------
// graph max pooling (gmp) via atomic max
__global__ void seg_pool_max(
    const float* __restrict__ X, const int* __restrict__ ibatch,
    float* __restrict__ out, int Nn, int C)
{
    long total = (long)Nn * C;
    long stride = (long)gridDim.x * blockDim.x;
    for (long i = blockIdx.x * (long)blockDim.x + threadIdx.x; i < total; i += stride) {
        long n = i / C;
        int  c = (int)(i % C);
        atomicMaxF(&out[(long)ibatch[n] * C + c], X[i]);
    }
}

// attention fusion over [xm,xg,xe]: logits = cat @ W^T + b, softmax, weighted sum
__global__ __launch_bounds__(128) void fuse_attn(
    const float* __restrict__ xm, const float* __restrict__ xg,
    const float* __restrict__ xe, const float* __restrict__ Wt, // [3, 3*OUTC]
    const float* __restrict__ bias, float* __restrict__ out, int C)
{
    int b = blockIdx.x, tid = threadIdx.x;
    __shared__ float red[3][128];
    __shared__ float wsm[3];
    float p0 = 0.f, p1 = 0.f, p2 = 0.f;
    int K3 = 3 * C;
    for (int j = tid; j < K3; j += 128) {
        float v = (j < C) ? xm[(long)b * C + j]
                : (j < 2 * C) ? xg[(long)b * C + j - C]
                              : xe[(long)b * C + j - 2 * C];
        p0 += Wt[j] * v;
        p1 += Wt[K3 + j] * v;
        p2 += Wt[2 * K3 + j] * v;
    }
    red[0][tid] = p0; red[1][tid] = p1; red[2][tid] = p2;
    __syncthreads();
    for (int o = 64; o; o >>= 1) {
        if (tid < o) {
            red[0][tid] += red[0][tid + o];
            red[1][tid] += red[1][tid + o];
            red[2][tid] += red[2][tid + o];
        }
        __syncthreads();
    }
    if (tid == 0) {
        float l0 = red[0][0] + bias[0];
        float l1 = red[1][0] + bias[1];
        float l2 = red[2][0] + bias[2];
        float mx = fmaxf(l0, fmaxf(l1, l2));
        float e0 = __expf(l0 - mx), e1 = __expf(l1 - mx), e2 = __expf(l2 - mx);
        float s = e0 + e1 + e2;
        wsm[0] = e0 / s; wsm[1] = e1 / s; wsm[2] = e2 / s;
    }
    __syncthreads();
    for (int c = tid; c < C; c += 128)
        out[(long)b * C + c] = wsm[0] * xm[(long)b * C + c]
                             + wsm[1] * xg[(long)b * C + c]
                             + wsm[2] * xe[(long)b * C + c];
}

// ---------------------------------------------------------------------------
// Host orchestration
extern "C" void kernel_launch(void* const* d_in, const int* in_sizes, int n_in,
                              void* d_out, int out_size, void* d_ws, size_t ws_size,
                              hipStream_t stream) {
    (void)in_sizes; (void)n_in; (void)out_size; (void)ws_size;
    // d_in layout assumption: setup_inputs() insertion order, with `params`
    // pytree-flattened by sorted keys (two levels, jax convention).
    auto P = [&](int i) { return (const float*)d_in[i]; };
    const float* DF   = P(0);
    const float* MUT  = P(1);
    const float* GEX  = P(2);
    const float* METH = P(3);
    const int*   ESRC = (const int*)d_in[76];
    const int*   EDST = ESRC + N_EDGES;
    const int*   IBAT = (const int*)d_in[77];
    float* OUT = (float*)d_out;

    const int iAttnW = 4, iAttnB = 5;
    const int iBnC1b = 6, iBnC1g = 7, iBnC2b = 8, iBnC2g = 9;
    const int iBnEndb = 10, iBnEndg = 11;
    const int iBnG1b = 12, iBnG1g = 13, iBnG2b = 14, iBnG2g = 15, iBnG3b = 16, iBnG3g = 17;
    const int iBnGe1b = 18, iBnGe1g = 19, iBnGe2b = 20, iBnGe2g = 21;
    const int iBnMe1b = 22, iBnMe1g = 23, iBnMe2b = 24, iBnMe2g = 25;
    const int iBnMu1b = 26, iBnMu1g = 27, iBnCb = 28, iBnCg = 29;
    const int iCov1W = 30, iCov1b = 31, iCov2W = 32, iCov2b = 33;
    const int iGe1W = 34, iGe1b = 35, iGe2W = 36, iGe2b = 37, iGe3W = 38, iGe3b = 39;
    const int iMe1W = 40, iMe1b = 41, iMe2W = 42, iMe2b = 43, iMe3W = 44, iMe3b = 45;
    const int iMu1W = 46, iMu1b = 47, iMu2W = 48, iMu2b = 49;
    const int iFcatW = 50, iFcatb = 51;
    const int iG1 = 52, iG2 = 58, iG3 = 64, iGE = 70; // +0 Wl +1 Wr +2 att +3 bias +4 bl +5 br

    // workspace layout (floats). Phase 1 (graph) and Phase 2 (omics) overlay.
    float* WSF = (float*)d_ws;
    float* XL   = WSF + 0L;
    float* XR   = WSF + 10240000L;
    float* HO   = WSF + 20480000L;
    float* X0   = WSF + 30720000L;
    float* X1   = WSF + 33280000L;
    float* TMP  = WSF + 35840000L;
    float* LOG_ = WSF + 38400000L;
    float* EXB  = WSF + 38760000L;
    float* SMAX = WSF + 39120000L;
    float* SDEN = WSF + 39160000L;
    float* C1   = WSF + 0L;          // phase 2 overlay (conv1 out, then im2col)
    float* IM2  = WSF + 0L;          // im2col matrix [98304 x 320] after P1 built
    float* P1   = WSF + 38043648L;
    float* C2   = WSF + 50724864L;   // conv2 gemm out [98304 x 32], channel-last
    float* P2   = WSF + 53870592L;
    float* XM1  = WSF + 54919168L;
    float* XM2  = WSF + 55181312L;
    float* XG1  = WSF + 55246848L;
    float* XG2  = WSF + 55508992L;
    float* XG3  = WSF + 55640064L;
    float* XE1  = WSF + 55705600L;
    float* XE2  = WSF + 55967744L;
    float* XE3  = WSF + 56098816L;
    float* FUS  = WSF + 56164352L;
    float* MEAN = WSF + 56229888L;
    float* VAR  = WSF + 56231936L;   // top ~56.23M floats (~215 MB)

    auto cdiv = [](long a, long b) { return (int)((a + b - 1) / b); };
    auto gemm = [&](const float* A, const float* W, const float* bias, float* Y,
                    int M, int N, int K, int act) {
        dim3 g(cdiv(M, BM), cdiv(N, BN));
        gemm_bias_act<<<g, 256, 0, stream>>>(A, W, bias, Y, M, N, K, act);
    };
    auto gemmT = [&](const float* A, const float* W, const float* bias, float* Y,
                     int M, int N, int K, int act) {
        dim3 g(cdiv(M, BM), cdiv(N, BN));
        gemm_tdm_bias_act<<<g, 256, 0, stream>>>(A, W, bias, Y, M, N, K, act);
    };
    auto fill = [&](float* p, float v, long n) {
        fill_f32<<<cdiv(n, 256), 256, 0, stream>>>(p, v, n);
    };
    auto bn1d = [&](const float* X, const float* res, int ib, int ig, float* Y,
                    long M, int N, int lin, int lout) {
        col_stats<<<N, 256, 0, stream>>>(X, (int)M, N, lin, MEAN, VAR);
        bn_apply<<<cdiv(M * N, 256), 256, 0, stream>>>(X, res, MEAN, VAR,
                                                       P(ig), P(ib), Y, M, N, lin, lout);
    };
    auto gat_layer = [&](const float* xin, int Kin, int H, int C, int ig,
                         int ibnB, int ibnG, const float* res, float* xout) {
        int HC = H * C;
        gemmT(xin, P(ig + 0), P(ig + 4), XL, N_NODES, HC, Kin, 0);
        gemmT(xin, P(ig + 1), P(ig + 5), XR, N_NODES, HC, Kin, 0);
        fill(SMAX, -3.0e38f, (long)N_NODES * H);
        fill(SDEN, 0.f, (long)N_NODES * H);
        fill(HO, 0.f, (long)N_NODES * HC);
        long waves = (long)E_TOT * H;
        edge_logits_max<<<cdiv(waves * 32, 256), 256, 0, stream>>>(
            XL, XR, P(ig + 2), ESRC, EDST, N_EDGES, E_TOT, H, C, LOG_, SMAX);
        edge_exp_den<<<cdiv(waves, 256), 256, 0, stream>>>(
            LOG_, SMAX, EDST, N_EDGES, E_TOT, H, EXB, SDEN);
        edge_scatter<<<cdiv(waves * 32, 256), 256, 0, stream>>>(
            XL, EXB, SDEN, ESRC, EDST, N_EDGES, E_TOT, H, C, HO);
        head_mean_bias<<<cdiv((long)N_NODES * C, 256), 256, 0, stream>>>(
            HO, P(ig + 3), TMP, N_NODES, H, C);
        bn1d(TMP, res, ibnB, ibnG, xout, N_NODES, C, /*lrelu_in=*/1, /*lrelu_out=*/0);
    };

    // ---------------- Phase 1: graph branch ----------------
    gat_layer(DF, F_IN, 4, UDIM, iG1, iBnG1b, iBnG1g, nullptr, X0);
    gat_layer(X0, UDIM, 4, UDIM, iG2, iBnG2b, iBnG2g, X0, X1);
    gat_layer(X1, UDIM, 4, UDIM, iG3, iBnG3b, iBnG3g, X1, X0);
    gat_layer(X0, UDIM, 1, OUTC, iGE, iBnEndb, iBnEndg, nullptr, X1); // X1: [10000,128]
    float* XDRUG = OUT + (long)BSZ * OUTC; // rows 512..1023 of output (pre final BN)
    fill(XDRUG, -3.0e38f, (long)NG * OUTC);
    seg_pool_max<<<cdiv((long)N_NODES * OUTC, 256), 256, 0, stream>>>(
        X1, IBAT, XDRUG, N_NODES, OUTC);

    // ---------------- Phase 2: mutation CNN ----------------
    const int W1 = (MUT_W - 9) / 3 + 1;  // 1161
    const int W1p = W1 / 3;              // 387
    const int W2 = (W1p - 5) / 2 + 1;    // 192
    const int W2p = W2 / 3;              // 64
    // conv1 (Ci=1, Kw=9): scalar; BN2d + lrelu; pool
    conv1d_k<<<cdiv((long)BSZ * 64 * W1, 256), 256, 0, stream>>>(
        MUT, P(iCov1W), P(iCov1b), C1, BSZ, 1, MUT_W, 64, 9, 3, W1);
    chan_stats<<<64, 256, 0, stream>>>(C1, BSZ, 64, W1, MEAN, VAR);
    bn2d_apply_lrelu<<<cdiv((long)BSZ * 64 * W1, 256), 256, 0, stream>>>(
        C1, MEAN, VAR, P(iBnC1g), P(iBnC1b), C1, BSZ, 64, W1);
    maxpool3_k<<<cdiv((long)BSZ * 64 * W1p, 256), 256, 0, stream>>>(C1, P1, BSZ * 64, W1, W1p);
    // conv2 (Ci=64, Kw=5, K=320) as im2col + WMMA GEMM, channel-last
    const int M2 = BSZ * W2;             // 98304 rows
    im2col_c2<<<cdiv((long)M2 * 320, 256), 256, 0, stream>>>(
        P1, IM2, BSZ, 64, W1p, 5, 2, W2);
    gemm(IM2, P(iCov2W), P(iCov2b), C2, M2, 32, 320, 0);
    // BN2d on channel-last layout == per-column BN over M2 rows; fuse lrelu
    col_stats<<<32, 256, 0, stream>>>(C2, M2, 32, 0, MEAN, VAR);
    bn_apply<<<cdiv((long)M2 * 32, 256), 256, 0, stream>>>(
        C2, nullptr, MEAN, VAR, P(iBnC2g), P(iBnC2b), C2, M2, 32, 0, 1);
    maxpool3_cl<<<cdiv((long)BSZ * 32 * W2p, 256), 256, 0, stream>>>(
        C2, P2, BSZ, 32, W2, W2p);
    // P2 is [512, 32*64=2048] flatten
    gemm(P2, P(iMu1W), P(iMu1b), XM1, BSZ, 512, 2048, 0);
    bn1d(XM1, nullptr, iBnMu1b, iBnMu1g, XM1, BSZ, 512, 0, 1);
    gemm(XM1, P(iMu2W), P(iMu2b), XM2, BSZ, OUTC, 512, 0);

    // ---------------- gexpr MLP ----------------
    gemm(GEX, P(iGe1W), P(iGe1b), XG1, BSZ, 512, DIM_GEXP, 0);
    bn1d(XG1, nullptr, iBnGe1b, iBnGe1g, XG1, BSZ, 512, 0, 1);
    gemm(XG1, P(iGe2W), P(iGe2b), XG2, BSZ, 256, 512, 0);
    bn1d(XG2, nullptr, iBnGe2b, iBnGe2g, XG2, BSZ, 256, 0, 1);
    gemm(XG2, P(iGe3W), P(iGe3b), XG3, BSZ, OUTC, 256, 0);

    // ---------------- methylation MLP ----------------
    gemm(METH, P(iMe1W), P(iMe1b), XE1, BSZ, 512, DIM_METH, 0);
    bn1d(XE1, nullptr, iBnMe1b, iBnMe1g, XE1, BSZ, 512, 0, 1);
    gemm(XE1, P(iMe2W), P(iMe2b), XE2, BSZ, 256, 512, 0);
    bn1d(XE2, nullptr, iBnMe2b, iBnMe2g, XE2, BSZ, 256, 0, 1);
    gemm(XE2, P(iMe3W), P(iMe3b), XE3, BSZ, OUTC, 256, 0);

    // ---------------- attention fusion + fcat ----------------
    fuse_attn<<<BSZ, 128, 0, stream>>>(XM2, XG3, XE3, P(iAttnW), P(iAttnB), FUS, OUTC);
    gemm(FUS, P(iFcatW), P(iFcatb), OUT, BSZ, OUTC, OUTC, /*lrelu=*/1); // rows 0..511

    // ---------------- final BN over concatenated [2B, OUT] ----------------
    col_stats<<<OUTC, 256, 0, stream>>>(OUT, 2 * BSZ, OUTC, 0, MEAN, VAR);
    bn_apply<<<cdiv((long)2 * BSZ * OUTC, 256), 256, 0, stream>>>(
        OUT, nullptr, MEAN, VAR, P(iBnCg), P(iBnCb), OUT, 2 * BSZ, OUTC, 0, 0);
}
